// CrossResonanceLayer_2061584302853
// MI455X (gfx1250) — matrix operand order repaired and checked by reading, not compiled, not run on hardware
//
#include <hip/hip_runtime.h>

typedef __bf16 bf16;
typedef __attribute__((ext_vector_type(16))) __bf16 v16bf;
typedef __attribute__((ext_vector_type(8)))  __bf16 bf16x8;
typedef __attribute__((ext_vector_type(8)))  float  v8f;
typedef __attribute__((ext_vector_type(4)))  unsigned u32x4;
typedef __attribute__((ext_vector_type(8)))  unsigned u32x8;

#define LSEQ   4096
#define LMASK  4095
#define DMODEL 512
#define BSZ    4
#define NROWS  (BSZ * LSEQ)   /* 16384 */
#define RADIUS 4
#define WIN    9

__device__ __forceinline__ bf16 f2bf(float f) {
  union { float f; unsigned u; } c; c.f = f;
  unsigned u = c.u;
  unsigned r = (u + 0x7FFFu + ((u >> 16) & 1u)) >> 16;   // round-to-nearest-even
  union { unsigned short s; bf16 b; } o; o.s = (unsigned short)r;
  return o.b;
}

// ---------------------------------------------------------------------------
// softmax(phase_weights) -> doubled tables (i in [0,8192)):
//   wbf[i]    = w[i & 4095]
//   winvbf[i] = w[(-i) & 4095]   (the reversed table)
// Identity used by the GEMMs: the *reversed* table of kernel w is winvbf, and
// the reversed table of kernel w_inv is wbf — so an ascending-index gather in
// the GEMM just swaps which table it is handed.
// ---------------------------------------------------------------------------
__global__ void prep_w_kernel(const float* __restrict__ pw,
                              bf16* __restrict__ wbf, bf16* __restrict__ winvbf) {
  __shared__ float sh[LSEQ];
  __shared__ float red[256];
  const int tid = threadIdx.x;
  float m = -1e30f;
  for (int j = 0; j < 16; ++j) m = fmaxf(m, pw[tid + j * 256]);
  red[tid] = m; __syncthreads();
  for (int s = 128; s > 0; s >>= 1) { if (tid < s) red[tid] = fmaxf(red[tid], red[tid + s]); __syncthreads(); }
  const float gmax = red[0]; __syncthreads();
  float lsum = 0.f;
  for (int j = 0; j < 16; ++j) {
    int i = tid + j * 256;
    float e = __expf(pw[i] - gmax);
    sh[i] = e; lsum += e;
  }
  red[tid] = lsum; __syncthreads();
  for (int s = 128; s > 0; s >>= 1) { if (tid < s) red[tid] += red[tid + s]; __syncthreads(); }
  const float inv = 1.0f / red[0];
  for (int j = 0; j < 16; ++j) {
    int i = tid + j * 256;
    bf16 wv  = f2bf(sh[i] * inv);
    bf16 wiv = f2bf(sh[(LSEQ - i) & LMASK] * inv);
    wbf[i]            = wv;  wbf[i + LSEQ]    = wv;
    winvbf[i]         = wiv; winvbf[i + LSEQ] = wiv;
  }
}

// ---------------------------------------------------------------------------
// f32 -> bf16 conversion (4 elems / thread)
// ---------------------------------------------------------------------------
__global__ void cvt_f32_bf16(const float* __restrict__ x, bf16* __restrict__ y, int n) {
  int i = (blockIdx.x * 256 + threadIdx.x) * 4;
  if (i + 3 < n) {
    float4 v = *(const float4*)&x[i];
    y[i]   = f2bf(v.x); y[i+1] = f2bf(v.y);
    y[i+2] = f2bf(v.z); y[i+3] = f2bf(v.w);
  }
}

// ---------------------------------------------------------------------------
// Circulant GEMM (FFT-conv equivalent):  Y[b,l,d] = sum_m w[(l-m)&4095] X[b,m,d]
// bf16 WMMA 16x16x32; wave -> 32x128 tile (16 accumulators, B-frag reused by
// two A-frags), block (8 waves) -> 256x128.
// `wrev` is the doubled *reversed* kernel table (wrev2[i] = w[(-i)&4095]),
// DMA'd into LDS by the Tensor Data Mover.  With the reversed table the
// A-gather is three contiguous ascending 16B DS loads per subtile — no
// reversal moves; the 2nd A-frag's high K-half aliases the 1st frag's low half.
// X slab double-buffered in LDS (ping-pong, one barrier per iteration).
// ---------------------------------------------------------------------------
template<bool WF32, bool WBF16>
__global__ void __launch_bounds__(256)
circulant_wmma(const bf16* __restrict__ wrev, const bf16* __restrict__ X,
               float* __restrict__ Yf, bf16* __restrict__ Ybf) {
  __shared__ __align__(16) bf16 wsh2[2 * LSEQ];   // doubled reversed table, 16KB
  __shared__ __align__(16) bf16 xsh[2][8192];     // [buf][ksub*4096 + col*32 + k]
  const int tid  = threadIdx.x;
  const int lane = tid & 31, wave = tid >> 5;
  const int n0 = blockIdx.x * 128;
  const int m0 = blockIdx.y * 256 + wave * 32;
  const long base = (long)blockIdx.z * LSEQ * DMODEL;

  // --- TDM: DMA the 16KB doubled table into LDS (wave 0 only)
  if (wave == 0) {
    unsigned long long ga = (unsigned long long)wrev;
    unsigned lds_off = (unsigned)(unsigned long long)wsh2;  // low 32b = LDS offset
    u32x4 g0;
    g0[0] = 1u;                                             // count=1, user mode
    g0[1] = lds_off;                                        // lds_addr
    g0[2] = (unsigned)(ga & 0xFFFFFFFFull);                 // global_addr[31:0]
    g0[3] = (unsigned)((ga >> 32) & 0x01FFFFFFull) | (2u << 30); // addr[56:32]|type=2
    u32x8 g1;
    g1[0] = (2u << 16);          // wg_mask=0, data_size=2 (4B dwords)
    g1[1] = (4096u << 16);       // tensor_dim0[15:0]=4096 dwords
    g1[2] = (1u << 16);          // tensor_dim0[31:16]=0, tensor_dim1=1
    g1[3] = (4096u << 16);       // tile_dim0=4096
    g1[4] = 1u;                  // tile_dim1=1, tile_dim2=0
    g1[5] = 4096u;               // tensor_dim0_stride
    g1[6] = 0u;
    g1[7] = 0u;
    asm volatile("tensor_load_to_lds %0, %1" :: "s"(g0), "s"(g1) : "memory");
    __builtin_amdgcn_s_wait_tensorcnt(0);
  }

  const int arow = m0 + (lane & 15);       // first A-frag row; second is arow+16
  const int kb   = (lane >> 4) * 8;        // 16-bit A layout: lane<16 -> K{0..7,16..23}
  const int col  = lane & 15;              // B-matrix column (N)
  const int kb2  = (lane >> 4) * 8;        // B layout: lane<16 -> K{0..7,16..23}, else K{8..15,24..31}
  const int ldr  = tid >> 2;               // staging: k row 0..63
  const int ldc  = (tid & 3) * 32;         // staging: col base 0,32,64,96
  const int lsub = (ldr >> 5) * 4096;      // which 32-k subtile
  const int lrr  = ldr & 31;
  const int gb   = kb - arow + LSEQ;       // ascending gather base (k0 added per sub)

  v8f acc0[8] = {}, acc1[8] = {};

  // prologue: stage k-slab 0 into buffer 0
  {
    const bf16* src = &X[base + (long)ldr * DMODEL + n0 + ldc];
    bf16x8 t0 = ((const bf16x8*)src)[0];
    bf16x8 t1 = ((const bf16x8*)src)[1];
    bf16x8 t2 = ((const bf16x8*)src)[2];
    bf16x8 t3 = ((const bf16x8*)src)[3];
#pragma unroll
    for (int j = 0; j < 8; ++j) {
      xsh[0][lsub + (ldc + j)      * 32 + lrr] = t0[j];
      xsh[0][lsub + (ldc + 8 + j)  * 32 + lrr] = t1[j];
      xsh[0][lsub + (ldc + 16 + j) * 32 + lrr] = t2[j];
      xsh[0][lsub + (ldc + 24 + j) * 32 + lrr] = t3[j];
    }
  }

  const int NKT = LSEQ / 64;   // 64 iterations
  for (int kt = 0; kt < NKT; ++kt) {
    __syncthreads();           // buf[kt&1] ready; prior reads of the other buf done
    const int cur = kt & 1;
    const bool more = (kt + 1) < NKT;
    bf16x8 t0, t1, t2, t3;
    if (more) {                // global loads overlap this iteration's WMMA burst
      const bf16* src = &X[base + (long)((kt + 1) * 64 + ldr) * DMODEL + n0 + ldc];
      t0 = ((const bf16x8*)src)[0];
      t1 = ((const bf16x8*)src)[1];
      t2 = ((const bf16x8*)src)[2];
      t3 = ((const bf16x8*)src)[3];
      if (kt + 2 < NKT)
        __builtin_prefetch(&X[base + (long)((kt + 2) * 64 + ldr) * DMODEL + n0 + ldc], 0, 3);
    }
#pragma unroll
    for (int sub = 0; sub < 2; ++sub) {
      const int ib = gb + kt * 64 + sub * 32;   // in [17, 8168]; single address VGPR
      v16bf a0, a1;
#pragma unroll
      for (int j = 0; j < 8; ++j) {
        a0[j]     = wsh2[ib + j];         // A0 K{kb..kb+7}       == w[arow-k0-kb-j]
        a0[j + 8] = wsh2[ib + 16 + j];    // A0 K{kb+16..kb+23}
        a1[j + 8] = a0[j];                // A1 high half == A0 low half (shift by 16)
        a1[j]     = wsh2[ib - 16 + j];    // A1 K{kb..kb+7} at row arow+16
      }
#pragma unroll
      for (int nt = 0; nt < 8; ++nt) {
        const bf16* bp = &xsh[cur][sub * 4096 + (nt * 16 + col) * 32 + kb2];
        bf16x8 blo = *(const bf16x8*)bp;
        bf16x8 bhi = *(const bf16x8*)(bp + 16);
        v16bf bb;
#pragma unroll
        for (int j = 0; j < 8; ++j) { bb[j] = blo[j]; bb[j + 8] = bhi[j]; }
        acc0[nt] = __builtin_amdgcn_wmma_f32_16x16x32_bf16(
            false, a0, false, bb, (short)0, acc0[nt], false, false);
        acc1[nt] = __builtin_amdgcn_wmma_f32_16x16x32_bf16(
            false, a1, false, bb, (short)0, acc1[nt], false, false);
      }
    }
    if (more) {
#pragma unroll
      for (int j = 0; j < 8; ++j) {
        xsh[1 - cur][lsub + (ldc + j)      * 32 + lrr] = t0[j];
        xsh[1 - cur][lsub + (ldc + 8 + j)  * 32 + lrr] = t1[j];
        xsh[1 - cur][lsub + (ldc + 16 + j) * 32 + lrr] = t2[j];
        xsh[1 - cur][lsub + (ldc + 24 + j) * 32 + lrr] = t3[j];
      }
    }
  }

  const int nlane = lane & 15;
  const int mbase = (lane >> 4) * 8;
#pragma unroll
  for (int nt = 0; nt < 8; ++nt)
#pragma unroll
    for (int r = 0; r < 8; ++r) {
      long off0 = base + (long)(m0 + mbase + r)      * DMODEL + n0 + nt * 16 + nlane;
      long off1 = base + (long)(m0 + 16 + mbase + r) * DMODEL + n0 + nt * 16 + nlane;
      float v0 = acc0[nt][r];
      float v1 = acc1[nt][r];
      if (WF32)  { Yf[off0]  = v0;       Yf[off1]  = v1; }
      if (WBF16) { Ybf[off0] = f2bf(v0); Ybf[off1] = f2bf(v1); }
    }
}

// ---------------------------------------------------------------------------
// Dense GEMM:  Y = X @ W^T (+bias)(+residual).  X is (16384 x 512) bf16,
// W is (512 x 512) f32 read row-major (== B-operand of the transposed product).
// wave -> 16x128 tile, block -> 128x128; W slab double-buffered (ping-pong).
// ---------------------------------------------------------------------------
template<bool BIAS, bool RESID, bool WF32, bool WBF16>
__global__ void __launch_bounds__(256)
dense_wmma_wt(const bf16* __restrict__ X, const float* __restrict__ Wt,
              const float* __restrict__ bias, const float* __restrict__ resid,
              float* __restrict__ Yf, bf16* __restrict__ Ybf) {
  __shared__ __align__(16) bf16 wsl[2][8192];   // [buf][ksub*4096 + n*32 + k]
  const int tid  = threadIdx.x;
  const int lane = tid & 31, wave = tid >> 5;
  const int n0 = blockIdx.x * 128;
  const int m0 = blockIdx.y * 128 + wave * 16;
  const int arow = m0 + (lane & 15);
  const int kb   = (lane >> 4) * 8;
  const int col  = lane & 15;
  const int kb2  = (lane >> 4) * 8;
  const int wn   = tid >> 1;              // 0..127 (n row)
  const int wk   = (tid & 1) * 32;        // k half: 0 or 32
  const int wsub = (tid & 1) * 4096;

  v8f acc[8] = {};

  // stage one 128n x 64k slab (f32 -> bf16) into buffer `buf`
  auto stage = [&](int buf, int kbase) {
    const float* wp = &Wt[(long)(n0 + wn) * DMODEL + kbase + wk];
    bf16x8 p[4];
#pragma unroll
    for (int i = 0; i < 4; ++i) {
      float4 fa = *(const float4*)(wp + i * 8);
      float4 fb = *(const float4*)(wp + i * 8 + 4);
      p[i][0]=f2bf(fa.x); p[i][1]=f2bf(fa.y); p[i][2]=f2bf(fa.z); p[i][3]=f2bf(fa.w);
      p[i][4]=f2bf(fb.x); p[i][5]=f2bf(fb.y); p[i][6]=f2bf(fb.z); p[i][7]=f2bf(fb.w);
    }
    bf16* d = &wsl[buf][wsub + wn * 32];
#pragma unroll
    for (int i = 0; i < 4; ++i) *(bf16x8*)(d + i * 8) = p[i];
  };

  stage(0, 0);

  const int NKT = DMODEL / 64;   // 8 iterations
  for (int kt = 0; kt < NKT; ++kt) {
    __syncthreads();
    const int cur = kt & 1;
    if (kt + 1 < NKT) stage(1 - cur, (kt + 1) * 64);
#pragma unroll
    for (int sub = 0; sub < 2; ++sub) {
      const int k0 = kt * 64 + sub * 32;
      bf16x8 alo = *(const bf16x8*)&X[(long)arow * DMODEL + k0 + kb];
      bf16x8 ahi = *(const bf16x8*)&X[(long)arow * DMODEL + k0 + kb + 16];
      v16bf a;
#pragma unroll
      for (int j = 0; j < 8; ++j) { a[j] = alo[j]; a[j + 8] = ahi[j]; }
#pragma unroll
      for (int nt = 0; nt < 8; ++nt) {
        const bf16* bp = &wsl[cur][sub * 4096 + (nt * 16 + col) * 32 + kb2];
        bf16x8 blo = *(const bf16x8*)bp;
        bf16x8 bhi = *(const bf16x8*)(bp + 16);
        v16bf bb;
#pragma unroll
        for (int j = 0; j < 8; ++j) { bb[j] = blo[j]; bb[j + 8] = bhi[j]; }
        acc[nt] = __builtin_amdgcn_wmma_f32_16x16x32_bf16(
            false, a, false, bb, (short)0, acc[nt], false, false);
      }
    }
  }

  const int nlane = lane & 15;
  const int mbase = (lane >> 4) * 8;
#pragma unroll
  for (int nt = 0; nt < 8; ++nt) {
    const int n = n0 + nt * 16 + nlane;
    float bv = BIAS ? bias[n] : 0.0f;
#pragma unroll
    for (int r = 0; r < 8; ++r) {
      long off = (long)(m0 + mbase + r) * DMODEL + n;
      float v = acc[nt][r] + bv;
      if (RESID) v += resid[off];
      if (WF32)  Yf[off]  = v;
      if (WBF16) Ybf[off] = f2bf(v);
    }
  }
}

// ---------------------------------------------------------------------------
// pooled = A.mean(axis=1)  -> (4, 512)
// ---------------------------------------------------------------------------
__global__ void pooled_mean(const float* __restrict__ A, float* __restrict__ pooled) {
  const int b = blockIdx.x;
  const int d = blockIdx.y * 256 + threadIdx.x;
  const float* p = A + (long)b * LSEQ * DMODEL + d;
  float s = 0.f;
  for (int l = 0; l < LSEQ; ++l) s += p[(long)l * DMODEL];
  pooled[b * DMODEL + d] = s * (1.0f / LSEQ);
}

// ---------------------------------------------------------------------------
// gate: 4x4 sims -> margin/entropy -> fire.any() scalar flag
// ---------------------------------------------------------------------------
__global__ void gate_kernel(const float* __restrict__ pooled, float* __restrict__ flag) {
  const int lane = threadIdx.x;  // 32 threads
  float sims[BSZ][BSZ];
  for (int i = 0; i < BSZ; ++i)
    for (int j = 0; j < BSZ; ++j) {
      float p = 0.f;
      for (int e = lane; e < DMODEL; e += 32)
        p += pooled[i * DMODEL + e] * pooled[j * DMODEL + e];
      for (int o = 16; o > 0; o >>= 1) p += __shfl_xor(p, o, 32);
      sims[i][j] = (i == j) ? p - 1e9f : p;
    }
  bool any = false;
  for (int i = 0; i < BSZ; ++i) {
    float t0 = -1e30f, t1 = -1e30f, mx = -1e30f;
    for (int j = 0; j < BSZ; ++j) {
      float v = sims[i][j];
      if (v > t0) { t1 = t0; t0 = v; } else if (v > t1) { t1 = v; }
      mx = fmaxf(mx, v);
    }
    float se = 0.f;
    for (int j = 0; j < BSZ; ++j) se += __expf(sims[i][j] - mx);
    float lse = __logf(se) + mx;
    float ent = 0.f;
    for (int j = 0; j < BSZ; ++j) {
      float lp = sims[i][j] - lse;
      float pr = __expf(lp);
      ent -= pr * __logf(pr + 1e-9f);
    }
    float conf = (t0 - t1) - 0.5f * ent;
    any = any || (conf < 0.15f);
  }
  if (lane == 0) flag[0] = any ? 1.0f : 0.0f;
}

// ---------------------------------------------------------------------------
// windowed local attention (W=9): one wave32 per (b,l) row.
// ---------------------------------------------------------------------------
__global__ void __launch_bounds__(256)
local_attn(const bf16* __restrict__ Q, const bf16* __restrict__ K,
           const bf16* __restrict__ V, bf16* __restrict__ ctx) {
  const int tid = threadIdx.x, lane = tid & 31, wave = tid >> 5;
  const int gl = blockIdx.x * 8 + wave;
  const int l  = gl & LMASK;
  const long rowbase = (long)(gl >> 12) * LSEQ * DMODEL;
  const int d0 = lane * 16;

  float q[16];
  {
    bf16x8 q0 = *(const bf16x8*)&Q[(long)gl * DMODEL + d0];
    bf16x8 q1 = *(const bf16x8*)&Q[(long)gl * DMODEL + d0 + 8];
#pragma unroll
    for (int j = 0; j < 8; ++j) { q[j] = (float)q0[j]; q[8 + j] = (float)q1[j]; }
  }

  float sc[WIN];
#pragma unroll
  for (int w = 0; w < WIN; ++w) {
    int row = (l + w - RADIUS) & LMASK;
    const bf16* kp = &K[rowbase + (long)row * DMODEL + d0];
    bf16x8 k0v = *(const bf16x8*)kp;
    bf16x8 k1v = *(const bf16x8*)(kp + 8);
    float p = 0.f;
#pragma unroll
    for (int j = 0; j < 8; ++j) p += q[j] * (float)k0v[j] + q[8 + j] * (float)k1v[j];
    for (int o = 16; o > 0; o >>= 1) p += __shfl_xor(p, o, 32);
    sc[w] = p * 0.04419417382415922f;   // 1/sqrt(512)
  }
  float mx = sc[0];
#pragma unroll
  for (int w = 1; w < WIN; ++w) mx = fmaxf(mx, sc[w]);
  float aw[WIN], se = 0.f;
#pragma unroll
  for (int w = 0; w < WIN; ++w) { aw[w] = __expf(sc[w] - mx); se += aw[w]; }
  const float inv = 1.0f / se;

  float c[16];
#pragma unroll
  for (int j = 0; j < 16; ++j) c[j] = 0.f;
#pragma unroll
  for (int w = 0; w < WIN; ++w) {
    int row = (l + w - RADIUS) & LMASK;
    const bf16* vp = &V[rowbase + (long)row * DMODEL + d0];
    bf16x8 v0 = *(const bf16x8*)vp;
    bf16x8 v1 = *(const bf16x8*)(vp + 8);
    float a = aw[w] * inv;
#pragma unroll
    for (int j = 0; j < 8; ++j) { c[j] += a * (float)v0[j]; c[8 + j] += a * (float)v1[j]; }
  }
#pragma unroll
  for (int j = 0; j < 16; ++j) ctx[(long)gl * DMODEL + d0 + j] = f2bf(c[j]);
}

// ---------------------------------------------------------------------------
// LayerNorm + gated select:  A_out = fire ? LN(h)*g+b : A   (h = A + rel)
// Emits both f32 (final output) and bf16 (input to second tuner).
// ---------------------------------------------------------------------------
__global__ void ln_select(const float* __restrict__ h, const float* __restrict__ A,
                          const float* __restrict__ g, const float* __restrict__ be,
                          const float* __restrict__ flag,
                          float* __restrict__ Aout, bf16* __restrict__ Aoutbf) {
  __shared__ float red[256];
  const int row = blockIdx.x, tid = threadIdx.x;
  const long base = (long)row * DMODEL;
  float x0 = h[base + tid], x1 = h[base + tid + 256];
  red[tid] = x0 + x1; __syncthreads();
  for (int s = 128; s > 0; s >>= 1) { if (tid < s) red[tid] += red[tid + s]; __syncthreads(); }
  const float mu = red[0] * (1.0f / DMODEL); __syncthreads();
  const float d0 = x0 - mu, d1 = x1 - mu;
  red[tid] = d0 * d0 + d1 * d1; __syncthreads();
  for (int s = 128; s > 0; s >>= 1) { if (tid < s) red[tid] += red[tid + s]; __syncthreads(); }
  const float rstd = rsqrtf(red[0] * (1.0f / DMODEL) + 1e-5f);
  const bool fire = flag[0] != 0.0f;
  float n0 = d0 * rstd * g[tid]       + be[tid];
  float n1 = d1 * rstd * g[tid + 256] + be[tid + 256];
  float o0 = fire ? n0 : A[base + tid];
  float o1 = fire ? n1 : A[base + tid + 256];
  Aout[base + tid]       = o0;
  Aout[base + tid + 256] = o1;
  Aoutbf[base + tid]       = f2bf(o0);
  Aoutbf[base + tid + 256] = f2bf(o1);
}

// ---------------------------------------------------------------------------
extern "C" void kernel_launch(void* const* d_in, const int* in_sizes, int n_in,
                              void* d_out, int out_size, void* d_ws, size_t ws_size,
                              hipStream_t stream) {
  const float* A   = (const float*)d_in[0];
  const float* B   = (const float*)d_in[1];
  const float* pw  = (const float*)d_in[2];
  const float* Wq  = (const float*)d_in[3];
  const float* Wk  = (const float*)d_in[4];
  const float* Wv  = (const float*)d_in[5];
  const float* Wo  = (const float*)d_in[6];
  const float* bo  = (const float*)d_in[7];
  const float* lns = (const float*)d_in[8];
  const float* lnb = (const float*)d_in[9];

  float* out  = (float*)d_out;
  const size_t NE = (size_t)NROWS * DMODEL;       // 8388608 per tensor
  float* Aout = out;
  float* Bout = out + NE;

  char* ws = (char*)d_ws;
  size_t off = 0;
  auto alloc = [&](size_t bytes) -> void* {
    void* p = ws + off;
    off += (bytes + 255) & ~(size_t)255;
    return p;
  };
  bf16*  wbf    = (bf16*) alloc(2 * LSEQ * 2);   // doubled w table
  bf16*  winvbf = (bf16*) alloc(2 * LSEQ * 2);   // doubled reversed (w_inv) table
  float* pooled = (float*)alloc(BSZ * DMODEL * 4);
  float* flag   = (float*)alloc(256);
  bf16*  Abf    = (bf16*) alloc(NE * 2);   // later reused as A_out bf16
  bf16*  Bbf    = (bf16*) alloc(NE * 2);   // later reused as ctx bf16
  bf16*  BalBf  = (bf16*) alloc(NE * 2);   // B_aligned bf16
  bf16*  Qbf    = (bf16*) alloc(NE * 2);
  bf16*  Kbf    = (bf16*) alloc(NE * 2);
  bf16*  Vbf    = (bf16*) alloc(NE * 2);
  float* hbuf   = Bout;                    // reuse d_out's B_out half as h scratch

  prep_w_kernel<<<1, 256, 0, stream>>>(pw, wbf, winvbf);
  cvt_f32_bf16<<<(int)(NE / 1024), 256, 0, stream>>>(A, Abf, (int)NE);
  cvt_f32_bf16<<<(int)(NE / 1024), 256, 0, stream>>>(B, Bbf, (int)NE);

  // tuner #1: kernel w  -> ascending gather uses its REVERSED table == winvbf
  circulant_wmma<false, true><<<dim3(4, 16, 4), 256, 0, stream>>>(winvbf, Bbf, nullptr, BalBf);

  pooled_mean<<<dim3(4, 2), 256, 0, stream>>>(A, pooled);
  gate_kernel<<<1, 32, 0, stream>>>(pooled, flag);

  // projections (K/V factored out of the 9-wide gather: 9x FLOP saving)
  dense_wmma_wt<false, false, false, true><<<dim3(4, 128), 256, 0, stream>>>(Abf,   Wq, nullptr, nullptr, nullptr, Qbf);
  dense_wmma_wt<false, false, false, true><<<dim3(4, 128), 256, 0, stream>>>(BalBf, Wk, nullptr, nullptr, nullptr, Kbf);
  dense_wmma_wt<false, false, false, true><<<dim3(4, 128), 256, 0, stream>>>(BalBf, Wv, nullptr, nullptr, nullptr, Vbf);

  local_attn<<<NROWS / 8, 256, 0, stream>>>(Qbf, Kbf, Vbf, Bbf /* ctx */);

  // h = A + ctx @ Wo^T + bo
  dense_wmma_wt<true, true, true, false><<<dim3(4, 128), 256, 0, stream>>>(Bbf, Wo, bo, A, hbuf, nullptr);

  // A_out = fire ? LN(h) : A   (f32 to d_out, bf16 for tuner #2)
  ln_select<<<NROWS, 256, 0, stream>>>(hbuf, A, lns, lnb, flag, Aout, Abf);

  // tuner #2: kernel w_inv -> ascending gather uses its REVERSED table == wbf
  circulant_wmma<true, false><<<dim3(4, 16, 4), 256, 0, stream>>>(wbf, Abf, Bout, nullptr);
}
